// mdLSTM_78503412236841
// MI455X (gfx1250) — matrix operand order repaired and checked by reading, not compiled, not run on hardware
//
#include <hip/hip_runtime.h>
#include <hip/hip_bf16.h>

typedef __attribute__((ext_vector_type(16))) _Float16 v16h;
typedef __attribute__((ext_vector_type(8)))  float    v8f;

// ---------------- problem dims ----------------
constexpr int Sn   = 4096;
constexpr int Tn   = 64;
constexpr int SPn  = 48;
constexpr int Pn   = 16;
constexpr int EMBn = 128;
constexpr int HIDn = 256;
constexpr int Gn   = 1024;   // 4*HID
constexpr int KTOT = 384;    // EMB + HID

// ---------------- ws layout (bytes) ----------------
constexpr size_t WS_LANEC = 0;                  // f32 [4096]
constexpr size_t WS_BIASG = 16384;              // f32 [1024]
constexpr size_t WS_WG    = 32768;              // f16 [1024][384] rows (async-staged)
constexpr size_t WS_OW    = 819200;             // f16 frag-order [8nt][8c][512]
constexpr size_t WS_FW1   = 884736;             // f16 frag-order [16nt][4c][512]
constexpr size_t WS_FW2   = 950272;             // f16 frag-order [4nt][8c][512]
constexpr size_t WS_EMB   = 1048576;            // f16 A-frag order [T][S/16][4c][512]
constexpr size_t WS_PRED  = 68157440;           // f16 A-frag order [4096rt][8c][512]
constexpr size_t WS_TMPH  = 101711872;          // f32 [S][T][HID] (optional)
constexpr size_t WS_TMPC  = 370147328;          // f32 [S][T][HID] (optional)
constexpr size_t WS_NEED_TMP = WS_TMPC + (size_t)Sn * Tn * HIDn * 4;  // 638,582,784

union FragU { v16h v; unsigned int u[8]; _Float16 h[16]; };

__device__ __forceinline__ v8f wmma16(v16h a, v16h b, v8f c) {
  return __builtin_amdgcn_wmma_f32_16x16x32_f16(false, a, false, b, (short)0, c,
                                                false, false);
}
__device__ __forceinline__ float sigf(float x) { return 1.f / (1.f + __expf(-x)); }
__device__ __forceinline__ v8f vzero() { return (v8f){0.f,0.f,0.f,0.f,0.f,0.f,0.f,0.f}; }

// ---------------- kernel 1: laneGate ----------------
__global__ void k_lanegate(const float* __restrict__ lane,
                           const float* __restrict__ lgW1, const float* __restrict__ lgb1,
                           const float* __restrict__ lgW2, const float* __restrict__ lgb2,
                           float* __restrict__ laneC) {
  int s = blockIdx.x * 256 + threadIdx.x;
  if (s >= Sn) return;
  float x = lane[s];
  float acc = lgb2[0];
#pragma unroll
  for (int i = 0; i < 32; ++i) {
    float h = x * lgW1[i] + lgb1[i];
    acc += (h > 0.f ? h : 0.f) * lgW2[i];
  }
  laneC[s] = 1.f / (1.f + __expf(-acc));
}

// ---------------- kernel 2: weight repack (LSTM rows + head frag-order) ----
__global__ void k_prepw(const float* __restrict__ Wih, const float* __restrict__ Whh,
                        const float* __restrict__ bih, const float* __restrict__ bhh,
                        const float* __restrict__ oW, const float* __restrict__ fW1,
                        const float* __restrict__ fW2,
                        _Float16* __restrict__ Wg, float* __restrict__ biasg,
                        _Float16* __restrict__ oWf, _Float16* __restrict__ f1f,
                        _Float16* __restrict__ f2f) {
  int gid = blockIdx.x * 256 + threadIdx.x;
  if (gid < Gn * KTOT) {                      // Wg[n][k] = {Wih[n][k] | Whh[n][k-128]}
    int n = gid / KTOT, kk = gid % KTOT;
    float v = (kk < EMBn) ? Wih[n * EMBn + kk] : Whh[n * HIDn + (kk - EMBn)];
    Wg[gid] = (_Float16)v;
  } else if (gid < 394240) {
    int n = gid - Gn * KTOT;
    biasg[n] = bih[n] + bhh[n];
  } else if (gid < 427008) {                  // oWf: B-frag order, K=256,N=128
    int i = gid - 394240;
    int nt = i >> 12, rem = i & 4095, c = rem >> 9, r2 = rem & 511;
    int ln = r2 >> 4, e = r2 & 15;
    int n = nt * 16 + (ln & 15), k = c * 32 + (e & 8) * 2 + (ln >> 4) * 8 + (e & 7);
    oWf[i] = (_Float16)oW[n * 256 + k];
  } else if (gid < 459776) {                  // f1f: K=128,N=256
    int i = gid - 427008;
    int nt = i >> 11, rem = i & 2047, c = rem >> 9, r2 = rem & 511;
    int ln = r2 >> 4, e = r2 & 15;
    int n = nt * 16 + (ln & 15), k = c * 32 + (e & 8) * 2 + (ln >> 4) * 8 + (e & 7);
    f1f[i] = (_Float16)fW1[n * 128 + k];
  } else if (gid < 476160) {                  // f2f: K=256,N=64
    int i = gid - 459776;
    int nt = i >> 12, rem = i & 4095, c = rem >> 9, r2 = rem & 511;
    int ln = r2 >> 4, e = r2 & 15;
    int n = nt * 16 + (ln & 15), k = c * 32 + (e & 8) * 2 + (ln >> 4) * 8 + (e & 7);
    f2f[i] = (_Float16)fW2[n * 256 + k];
  }
}

// ---------------- kernel 3: embedding MLP -> A-frag-order f16 ----------------
__global__ void __launch_bounds__(256) k_embed(const float* __restrict__ inp,
                                               const float* __restrict__ laneC,
                                               const float* __restrict__ eW1,
                                               const float* __restrict__ eb1,
                                               const float* __restrict__ eW2,
                                               const float* __restrict__ eb2,
                                               _Float16* __restrict__ emb) {
  __shared__ float sW1[64 * 3], sb1[64], sW2[128 * 64], sb2[128];
  int tid = threadIdx.x;
  for (int i = tid; i < 192; i += 256) sW1[i] = eW1[i];
  for (int i = tid; i < 64; i += 256) sb1[i] = eb1[i];
  for (int i = tid; i < 8192; i += 256) sW2[i] = eW2[i];
  for (int i = tid; i < 128; i += 256) sb2[i] = eb2[i];
  __syncthreads();

  int gid = blockIdx.x * 256 + tid;           // [0, S*T)
  int t = gid >> 12, s = gid & (Sn - 1);
  float lc = laneC[s];
  const float* xp = inp + (size_t)s * (Tn * 3) + t * 3;
  float x0 = xp[0] * lc, x1 = xp[1] * lc, x2 = xp[2] * lc;
  float h[64];
#pragma unroll
  for (int i = 0; i < 64; ++i) {
    float v = sW1[i * 3] * x0 + sW1[i * 3 + 1] * x1 + sW1[i * 3 + 2] * x2 + sb1[i];
    h[i] = v > 0.f ? v : 0.f;
  }
  const int stile = s >> 4, m = s & 15;
  _Float16* base = emb + (((size_t)t * 256 + stile) * 4) * 512;
#pragma unroll
  for (int c = 0; c < 4; ++c) {
#pragma unroll
    for (int hs2 = 0; hs2 < 2; ++hs2) {
      FragU g;
#pragma unroll
      for (int e = 0; e < 16; ++e) {
        int j = c * 32 + ((e < 8) ? (hs2 * 8 + e) : (16 + hs2 * 8 + (e - 8)));
        float a = sb2[j];
        const float* wr = sW2 + j * 64;
#pragma unroll 8
        for (int i2 = 0; i2 < 64; ++i2) a += wr[i2] * h[i2];
        g.h[e] = (_Float16)(a > 0.f ? a : 0.f);
      }
      *(v16h*)(base + (size_t)c * 512 + (hs2 * 16 + m) * 16) = g.v;
    }
  }
}

// ---------------- kernel 4: sequential LSTM with async-staged WMMA ----------
constexpr unsigned LBUF0 = 0, LBUF1 = 65536;
constexpr unsigned LS_C = 131072;
constexpr unsigned LS_HF = 163840;
constexpr size_t   LSTM_LDS = 180224;

// 16 async B128 pieces per thread; constant strides (LDS +4096B, global +49152B)
// => 2 live address registers instead of 16 unrolled pairs.
__device__ __forceinline__ void issue_chunk_async(const _Float16* __restrict__ Wg,
                                                  unsigned ldsDstBase, int kb, int tid) {
  const int n0 = tid >> 2, p = tid & 3;
  unsigned lds = ldsDstBase +
      ((unsigned)(n0 >> 4) * 512u +
       (unsigned)(((p >> 1) * 16 + (n0 & 15)) * 16) +
       (unsigned)((p & 1) * 8)) * 2u;
  unsigned long long ga =
      (unsigned long long)(Wg + (size_t)n0 * KTOT + kb + (p & 1) * 16 + (p >> 1) * 8);
  for (int i = 0; i < 16; ++i) {
    asm volatile("global_load_async_to_lds_b128 %0, %1, off"
                 :: "v"(lds), "v"(ga) : "memory");
    lds += 4096u;                              // +4 tiles * 1024B
    ga  += (unsigned long long)(64 * KTOT * 2);// +64 rows
  }
}

__global__ void __launch_bounds__(256) k_lstm(const _Float16* __restrict__ emb,
                                              const _Float16* __restrict__ Wg,
                                              const float* __restrict__ biasg,
                                              const float* __restrict__ convw,
                                              const float* __restrict__ convb,
                                              float* __restrict__ Hout,
                                              float* __restrict__ Cout,
                                              float* __restrict__ tmpH,
                                              float* __restrict__ tmpC,
                                              _Float16* __restrict__ pred,
                                              int useTmp) {
  extern __shared__ char smem[];
  _Float16* lsBuf0 = (_Float16*)(smem + LBUF0);
  _Float16* lsBuf1 = (_Float16*)(smem + LBUF1);
  float*    lsG    = (float*)smem;            // union with buffers
  float*    lsC    = (float*)(smem + LS_C);
  _Float16* lsHf   = (_Float16*)(smem + LS_HF);
  const unsigned smemBase = (unsigned)(size_t)smem;

  const int tid = threadIdx.x;
  const int lane = tid & 31;
  const int w  = tid >> 5;
  const int mg = w >> 2;
  const int ng = w & 3;
  const int nlo = lane & 15;
  const int hs  = lane >> 4;
  const int s0 = blockIdx.x * 32;

  for (int i = tid; i < 32 * HIDn; i += 256) lsC[i] = 0.f;
  for (int i = tid; i < 2 * 8 * 512; i += 256) lsHf[i] = (_Float16)0.f;

  float bg[16];
#pragma unroll
  for (int nt = 0; nt < 16; ++nt) bg[nt] = biasg[ng * 256 + nt * 16 + nlo];
  const float cw0 = convw[0], cw1 = convw[1], cw2 = convw[2], cb = convb[0];

  const int jj = tid;                         // elementwise column per thread
  const int cj = jj >> 5, koffj = jj & 31;
  const int hsj = (koffj >> 3) & 1;
  const int ej  = (koffj & 7) + 8 * (koffj >> 4);
  __syncthreads();

  for (int k = 0; k < 65; ++k) {
    const int t = (k <= SPn) ? k : (k - 1);
    const int slot = (k < SPn) ? k : (k - 1);

    v8f acc[16];
#pragma unroll
    for (int nt = 0; nt < 16; ++nt) acc[nt] = vzero();

    issue_chunk_async(Wg, smemBase + LBUF0, 0, tid);
    for (int kc = 0; kc < 12; ++kc) {
      __syncthreads();                        // consumers of target buffer done
      // A fragment: step-stable data -> overlap with async wait
      v16h a;
      if (kc < 4)
        a = *(const v16h*)(emb + ((((size_t)t * 256 + (s0 >> 4) + mg) * 4) + kc) * 512
                           + lane * 16);
      else
        a = *(const v16h*)(lsHf + ((size_t)(mg * 8 + (kc - 4))) * 512 + lane * 16);

      if (kc < 11) {
        issue_chunk_async(Wg, smemBase + ((kc & 1) ? LBUF0 : LBUF1), (kc + 1) * 32, tid);
        asm volatile("s_wait_asynccnt 0x10" ::: "memory");
      } else {
        asm volatile("s_wait_asynccnt 0x0" ::: "memory");
      }
      __syncthreads();                        // chunk kc visible to all

      const _Float16* bp = ((kc & 1) ? lsBuf1 : lsBuf0) + (size_t)(ng * 16) * 512
                           + lane * 16;
      // software-pipelined B fragments: load nt+1 while WMMA nt runs
      v16h bc = *(const v16h*)bp;
#pragma unroll
      for (int nt = 0; nt < 15; ++nt) {
        v16h bn = *(const v16h*)(bp + (size_t)(nt + 1) * 512);
        acc[nt] = wmma16(a, bc, acc[nt]);
        bc = bn;
      }
      acc[15] = wmma16(a, bc, acc[15]);
    }
    __syncthreads();

    // gates (+bias) -> LDS f32 [32][1024]
#pragma unroll
    for (int nt = 0; nt < 16; ++nt) {
      int col = ng * 256 + nt * 16 + nlo;
#pragma unroll
      for (int e = 0; e < 8; ++e) {
        int row = mg * 16 + e + 8 * hs;
        lsG[row * Gn + col] = acc[nt][e] + bg[nt];
      }
    }
    __syncthreads();

    // elementwise cell update (thread = column jj, loop rows)
    for (int r = 0; r < 32; ++r) {
      float gi = lsG[r * Gn + jj];
      float gf = lsG[r * Gn + 256 + jj];
      float gg = lsG[r * Gn + 512 + jj];
      float go = lsG[r * Gn + 768 + jj];
      float c = lsC[r * HIDn + jj];
      float cn = sigf(gf) * c + sigf(gi) * tanhf(gg);
      lsC[r * HIDn + jj] = cn;
      lsG[r * Gn + jj] = sigf(go) * tanhf(cn);   // hraw over i-gate slot
      if (k != SPn) {
        if (useTmp) tmpC[(((size_t)(s0 + r)) * Tn + slot) * HIDn + jj] = cn;
        else Cout[(size_t)(s0 + r) * (HIDn * Tn) + (size_t)jj * Tn + slot] = cn;
      }
    }
    __syncthreads();

    // conv over hidden dim; write h (A-frag LDS, outputs, pred frags)
    for (int r = 0; r < 32; ++r) {
      float hm = (jj > 0)   ? lsG[r * Gn + jj - 1] : 0.f;
      float h0 = lsG[r * Gn + jj];
      float hp = (jj < 255) ? lsG[r * Gn + jj + 1] : 0.f;
      float h = cw0 * hm + cw1 * h0 + cw2 * hp + cb;
      lsHf[(((r >> 4) * 8 + cj) * 32 + hsj * 16 + (r & 15)) * 16 + ej] = (_Float16)h;
      if (k != SPn) {
        if (useTmp) tmpH[(((size_t)(s0 + r)) * Tn + slot) * HIDn + jj] = h;
        else Hout[(size_t)(s0 + r) * (HIDn * Tn) + (size_t)jj * Tn + slot] = h;
      }
      if (k >= SPn + 1) {
        int p = k - (SPn + 1);
        pred[(((size_t)(s0 + r)) * 8 + cj) * 512 + (hsj * 16 + p) * 16 + ej] =
            (_Float16)h;
      }
    }
    __syncthreads();
  }
}

// ---------------- kernel 5: [S][T][HID] -> [S][HID][T] tiled transpose -------
__global__ void __launch_bounds__(256) k_xpose(const float* __restrict__ tmp,
                                               float* __restrict__ outp) {
  extern __shared__ char smem[];
  float* tile = (float*)smem;                 // [64][257]
  const size_t s = blockIdx.x;
  const float* src = tmp + s * (size_t)(Tn * HIDn);
  float* dst = outp + s * (size_t)(Tn * HIDn);
  const int tid = threadIdx.x;
  for (int idx = tid; idx < Tn * HIDn; idx += 256) {
    int t = idx >> 8, j = idx & 255;
    tile[t * 257 + j] = src[idx];
  }
  __syncthreads();
  float4* d4 = (float4*)(dst + (size_t)tid * Tn);
#pragma unroll
  for (int t4 = 0; t4 < 16; ++t4) {
    float4 v;
    v.x = tile[(t4 * 4 + 0) * 257 + tid];
    v.y = tile[(t4 * 4 + 1) * 257 + tid];
    v.z = tile[(t4 * 4 + 2) * 257 + tid];
    v.w = tile[(t4 * 4 + 3) * 257 + tid];
    d4[t4] = v;
  }
}

// ---------------- kernel 6: head (weights frag-order in L2, acts in LDS) ----
__global__ void __launch_bounds__(128) k_head(const _Float16* __restrict__ pred,
                                              const _Float16* __restrict__ oWf,
                                              const _Float16* __restrict__ f1f,
                                              const _Float16* __restrict__ f2f,
                                              const float* __restrict__ ob,
                                              const float* __restrict__ fb1,
                                              const float* __restrict__ fb2,
                                              const float* __restrict__ fW3,
                                              const float* __restrict__ fb3,
                                              const float* __restrict__ laneC,
                                              float* __restrict__ out) {
  __shared__ _Float16 acts[4 * 7168];
  const int tid = threadIdx.x, lane = tid & 31, w = tid >> 5;
  const int nlo = lane & 15, hs = lane >> 4;
  _Float16* act1 = acts + w * 7168;           // [4c][512]
  _Float16* act2 = act1 + 2048;               // [8c][512]
  _Float16* act3 = act2 + 4096;               // [2c][512]
  const int rt = blockIdx.x * 4 + w;          // 16-row tile id

  // L0: 256 -> 128, +ob, no relu
  {
    v8f acc[8];
#pragma unroll
    for (int nt = 0; nt < 8; ++nt) acc[nt] = vzero();
    for (int kc = 0; kc < 8; ++kc) {
      v16h a = *(const v16h*)(pred + ((size_t)rt * 8 + kc) * 512 + lane * 16);
      if (kc < 7)
        __builtin_prefetch((const void*)(pred + ((size_t)rt * 8 + kc + 1) * 512), 0, 1);
      const _Float16* bp = oWf + (size_t)kc * 512 + lane * 16;
      v16h bc = *(const v16h*)bp;
#pragma unroll
      for (int nt = 0; nt < 7; ++nt) {
        v16h bn = *(const v16h*)(bp + (size_t)(nt + 1) * 4096);
        acc[nt] = wmma16(a, bc, acc[nt]);
        bc = bn;
      }
      acc[7] = wmma16(a, bc, acc[7]);
    }
#pragma unroll
    for (int nt = 0; nt < 8; ++nt) {
      float bb = ob[nt * 16 + nlo];
      int c = nt >> 1, e2 = (nlo & 7) + 8 * (nt & 1), hs2 = (nlo >> 3) & 1;
#pragma unroll
      for (int e = 0; e < 8; ++e)
        act1[c * 512 + (hs2 * 16 + e + 8 * hs) * 16 + e2] = (_Float16)(acc[nt][e] + bb);
    }
  }
  __syncthreads();

  // L1: 128 -> 256, +fb1, relu
  {
    v8f acc[16];
#pragma unroll
    for (int nt = 0; nt < 16; ++nt) acc[nt] = vzero();
    for (int kc = 0; kc < 4; ++kc) {
      v16h a = *(const v16h*)(act1 + (size_t)kc * 512 + lane * 16);
      const _Float16* bp = f1f + (size_t)kc * 512 + lane * 16;
      v16h bc = *(const v16h*)bp;
#pragma unroll
      for (int nt = 0; nt < 15; ++nt) {
        v16h bn = *(const v16h*)(bp + (size_t)(nt + 1) * 2048);
        acc[nt] = wmma16(a, bc, acc[nt]);
        bc = bn;
      }
      acc[15] = wmma16(a, bc, acc[15]);
    }
#pragma unroll
    for (int nt = 0; nt < 16; ++nt) {
      float bb = fb1[nt * 16 + nlo];
      int c = nt >> 1, e2 = (nlo & 7) + 8 * (nt & 1), hs2 = (nlo >> 3) & 1;
#pragma unroll
      for (int e = 0; e < 8; ++e) {
        float v = acc[nt][e] + bb;
        act2[c * 512 + (hs2 * 16 + e + 8 * hs) * 16 + e2] = (_Float16)(v > 0.f ? v : 0.f);
      }
    }
  }
  __syncthreads();

  // L2: 256 -> 64, +fb2, relu
  {
    v8f acc[4];
#pragma unroll
    for (int nt = 0; nt < 4; ++nt) acc[nt] = vzero();
    for (int kc = 0; kc < 8; ++kc) {
      v16h a = *(const v16h*)(act2 + (size_t)kc * 512 + lane * 16);
      const _Float16* bp = f2f + (size_t)kc * 512 + lane * 16;
      v16h bc = *(const v16h*)bp;
#pragma unroll
      for (int nt = 0; nt < 3; ++nt) {
        v16h bn = *(const v16h*)(bp + (size_t)(nt + 1) * 4096);
        acc[nt] = wmma16(a, bc, acc[nt]);
        bc = bn;
      }
      acc[3] = wmma16(a, bc, acc[3]);
    }
#pragma unroll
    for (int nt = 0; nt < 4; ++nt) {
      float bb = fb2[nt * 16 + nlo];
      int c = nt >> 1, e2 = (nlo & 7) + 8 * (nt & 1), hs2 = (nlo >> 3) & 1;
#pragma unroll
      for (int e = 0; e < 8; ++e) {
        float v = acc[nt][e] + bb;
        act3[c * 512 + (hs2 * 16 + e + 8 * hs) * 16 + e2] = (_Float16)(v > 0.f ? v : 0.f);
      }
    }
  }
  __syncthreads();

  // L3: dot64 + fb3, / laneC
  if (lane < 16) {
    float sacc = fb3[0];
#pragma unroll
    for (int kk = 0; kk < 64; ++kk) {
      int c = kk >> 5, koff = kk & 31;
      int h2 = (koff >> 3) & 1, e = (koff & 7) + 8 * (koff >> 4);
      sacc += (float)act3[c * 512 + (h2 * 16 + lane) * 16 + e] * fW3[kk];
    }
    int row = rt * 16 + lane;
    out[row] = sacc / laneC[row >> 4];
  }
}

// ---------------- launcher ----------------
extern "C" void kernel_launch(void* const* d_in, const int* in_sizes, int n_in,
                              void* d_out, int out_size, void* d_ws, size_t ws_size,
                              hipStream_t stream) {
  const float* inputData = (const float*)d_in[0];
  const float* lane  = (const float*)d_in[1];
  const float* lgW1  = (const float*)d_in[2];
  const float* lgb1  = (const float*)d_in[3];
  const float* lgW2  = (const float*)d_in[4];
  const float* lgb2  = (const float*)d_in[5];
  const float* eW1   = (const float*)d_in[6];
  const float* eb1   = (const float*)d_in[7];
  const float* eW2   = (const float*)d_in[8];
  const float* eb2   = (const float*)d_in[9];
  const float* Wih   = (const float*)d_in[10];
  const float* Whh   = (const float*)d_in[11];
  const float* bih   = (const float*)d_in[12];
  const float* bhh   = (const float*)d_in[13];
  const float* convw = (const float*)d_in[14];
  const float* convb = (const float*)d_in[15];
  const float* oW    = (const float*)d_in[16];
  const float* ob    = (const float*)d_in[17];
  const float* fW1   = (const float*)d_in[18];
  const float* fb1   = (const float*)d_in[19];
  const float* fW2   = (const float*)d_in[20];
  const float* fb2   = (const float*)d_in[21];
  const float* fW3   = (const float*)d_in[22];
  const float* fb3   = (const float*)d_in[23];

  char* ws = (char*)d_ws;
  float*    laneC = (float*)(ws + WS_LANEC);
  float*    biasg = (float*)(ws + WS_BIASG);
  _Float16* Wg    = (_Float16*)(ws + WS_WG);
  _Float16* oWf   = (_Float16*)(ws + WS_OW);
  _Float16* f1f   = (_Float16*)(ws + WS_FW1);
  _Float16* f2f   = (_Float16*)(ws + WS_FW2);
  _Float16* emb16 = (_Float16*)(ws + WS_EMB);
  _Float16* pred16= (_Float16*)(ws + WS_PRED);
  float*    tmpH  = (float*)(ws + WS_TMPH);
  float*    tmpC  = (float*)(ws + WS_TMPC);
  const int useTmp = (ws_size >= WS_NEED_TMP) ? 1 : 0;

  float* outP = (float*)d_out;
  float* Hout = outP + (size_t)Sn * Pn;
  float* Cout = Hout + (size_t)Sn * HIDn * Tn;

  k_lanegate<<<16, 256, 0, stream>>>(lane, lgW1, lgb1, lgW2, lgb2, laneC);
  k_prepw<<<1860, 256, 0, stream>>>(Wih, Whh, bih, bhh, oW, fW1, fW2,
                                    Wg, biasg, oWf, f1f, f2f);
  k_embed<<<(Sn * Tn) / 256, 256, 0, stream>>>(inputData, laneC, eW1, eb1, eW2, eb2, emb16);

  (void)hipFuncSetAttribute(reinterpret_cast<const void*>(k_lstm),
                            hipFuncAttributeMaxDynamicSharedMemorySize, (int)LSTM_LDS);
  k_lstm<<<Sn / 32, 256, LSTM_LDS, stream>>>(emb16, Wg, biasg, convw, convb,
                                             Hout, Cout, tmpH, tmpC, pred16, useTmp);
  if (useTmp) {
    (void)hipFuncSetAttribute(reinterpret_cast<const void*>(k_xpose),
                              hipFuncAttributeMaxDynamicSharedMemorySize, 65792);
    k_xpose<<<Sn, 256, 65792, stream>>>(tmpH, Hout);
    k_xpose<<<Sn, 256, 65792, stream>>>(tmpC, Cout);
  }
  k_head<<<(Sn * Pn) / 64, 128, 0, stream>>>(pred16, oWf, f1f, f2f,
                                             ob, fb1, fb2, fW3, fb3, laneC, outP);
}
